// AttentionBlock_31834297598302
// MI455X (gfx1250) — matrix operand order repaired and checked by reading, not compiled, not run on hardware
//
#include <hip/hip_runtime.h>

// ---------------------------------------------------------------------------
// CDNA5 (gfx1250) attention block: prep (bf16 convert/transpose) -> QKV proj
// -> flash attention -> out proj.  All GEMMs via v_wmma_f32_16x16x32_bf16.
// All operands laid out K-contiguous so every fragment = two b128 loads.
// One wave32 per (blocked) output tile; EXEC all ones (WMMA requirement).
// ---------------------------------------------------------------------------

typedef __attribute__((ext_vector_type(16))) __bf16         v16bf;
typedef __attribute__((ext_vector_type(16))) unsigned short v16u;
typedef __attribute__((ext_vector_type(8)))  unsigned short v8us;
typedef __attribute__((ext_vector_type(8)))  float          v8f;

union Frag { v16u u; v16bf b; v8us h[2]; };

#define BN   8
#define CN   256
#define SN   2048
#define HN   4
#define DKN  64
#define HDN  256   /* H*dk */
#define ON   768   /* 3*H*dk */

__device__ __forceinline__ unsigned short f2bf(float f) {
  union { float f; unsigned u; } c; c.f = f;
  return (unsigned short)((c.u + 0x7FFFu + ((c.u >> 16) & 1u)) >> 16);
}

// A-fragment (16x32 bf16 MxK) for row pointer p (K-contiguous):
//   elems 0..7  = p[hi*8   .. hi*8+7]
//   elems 8..15 = p[16+hi*8 .. 16+hi*8+7]        (ISA 7.12.2)
__device__ __forceinline__ Frag loadA(const unsigned short* p, int hi) {
  Frag f;
  f.h[0] = *(const v8us*)(p + hi * 8);
  f.h[1] = *(const v8us*)(p + 16 + hi * 8);
  return f;
}
// B-fragment (32x16 bf16 KxN) for column pointer p (K-contiguous):
//   elems 0..7  = p[hi*8   .. hi*8+7]
//   elems 8..15 = p[16+hi*8 .. 16+hi*8+7]
__device__ __forceinline__ Frag loadB(const unsigned short* p, int hi) {
  Frag f;
  f.h[0] = *(const v8us*)(p + hi * 8);
  f.h[1] = *(const v8us*)(p + 16 + hi * 8);
  return f;
}

__device__ __forceinline__ v8f wmma_bf16(Frag a, Frag b, v8f c) {
  return __builtin_amdgcn_wmma_f32_16x16x32_bf16(false, a.b, false, b.b,
                                                 (short)0, c, false, false);
}

// ---------------------------------------------------------------------------
// Prep: fp32 -> bf16 elementwise (weights)
// ---------------------------------------------------------------------------
__global__ __launch_bounds__(256) void cvt_kernel(const float* __restrict__ src,
                                                  unsigned short* __restrict__ dst,
                                                  int n) {
  int i = blockIdx.x * 256 + threadIdx.x;
  if (i < n) dst[i] = f2bf(src[i]);
}

// ---------------------------------------------------------------------------
// Prep: xbf[b][s][c] = bf16(x[b][c][s])  (LDS-tiled transpose, coalesced)
// ---------------------------------------------------------------------------
__global__ __launch_bounds__(256) void xt_kernel(const float* __restrict__ x,
                                                 unsigned short* __restrict__ xbf) {
  __shared__ float tile[32][33];
  const int bb = blockIdx.z;
  const int s0 = blockIdx.x * 32, c0 = blockIdx.y * 32;
  const int tx = threadIdx.x, ty = threadIdx.y;  // (32,8)
  const float* xb = x + (size_t)bb * CN * SN;
#pragma unroll
  for (int j = 0; j < 32; j += 8)
    tile[ty + j][tx] = xb[(size_t)(c0 + ty + j) * SN + s0 + tx];
  __syncthreads();
  unsigned short* ob = xbf + (size_t)bb * SN * CN;
#pragma unroll
  for (int j = 0; j < 32; j += 8)
    ob[(size_t)(s0 + ty + j) * CN + c0 + tx] = f2bf(tile[tx][ty + j]);
}

// ---------------------------------------------------------------------------
// Kernel 1: qkv[b,s,o] = sum_c xbf[b,s,c]*wpb[o,c] + b_proj[o]
// 16(s) x 32(o) per wave (A reused for 2 B tiles).
// q,k -> [b][h][s][d]; v -> transposed [b][h][d][s] (for contiguous PV frags).
// ---------------------------------------------------------------------------
__global__ __launch_bounds__(32) void qkv_kernel(
    const unsigned short* __restrict__ xbf, const unsigned short* __restrict__ wpb,
    const float* __restrict__ b_proj,
    unsigned short* __restrict__ qw, unsigned short* __restrict__ kw,
    unsigned short* __restrict__ vtw) {
  const int lane = threadIdx.x, hi = lane >> 4, r = lane & 15;
  const int s0 = blockIdx.x * 16, o0 = blockIdx.y * 32, bb = blockIdx.z;

  const unsigned short* arow = xbf + ((size_t)bb * SN + s0 + r) * CN;
  const unsigned short* brow0 = wpb + (size_t)(o0 + r) * CN;
  const unsigned short* brow1 = wpb + (size_t)(o0 + 16 + r) * CN;

  v8f acc0 = {}, acc1 = {};
  for (int kk = 0; kk < CN; kk += 32) {
    Frag a  = loadA(arow + kk, hi);
    Frag b0 = loadB(brow0 + kk, hi);
    Frag b1 = loadB(brow1 + kk, hi);
    acc0 = wmma_bf16(a, b0, acc0);
    acc1 = wmma_bf16(a, b1, acc1);
  }
#pragma unroll
  for (int ot = 0; ot < 2; ++ot) {
    const v8f acc = ot ? acc1 : acc0;
    const int o = o0 + ot * 16 + r;
    const int h = o / 192, t = (o % 192) / 64, d = o % 64;
#pragma unroll
    for (int i = 0; i < 8; ++i) {
      const int s = s0 + i + hi * 8;
      const unsigned short bf = f2bf(acc[i] + b_proj[o]);
      if (t == 0)
        qw[(((size_t)bb * HN + h) * SN + s) * DKN + d] = bf;
      else if (t == 1)
        kw[(((size_t)bb * HN + h) * SN + s) * DKN + d] = bf;
      else
        vtw[(((size_t)bb * HN + h) * DKN + d) * SN + s] = bf;
    }
  }
}

// ---------------------------------------------------------------------------
// Kernel 2: flash attention per (b,h).  32 queries per wave (2 tiles sharing
// every K/V fragment), 32-key blocks: 8 WMMA for S, 8 WMMA for PV per block.
// ---------------------------------------------------------------------------
__global__ __launch_bounds__(32) void attn_kernel(
    const unsigned short* __restrict__ q, const unsigned short* __restrict__ k,
    const unsigned short* __restrict__ vT, unsigned short* __restrict__ res) {
  __shared__ __align__(16) unsigned short p_lds[2][16 * 32];

  const int lane = threadIdx.x, hi = lane >> 4, r = lane & 15;
  const int s0 = blockIdx.x * 32;
  const int bh = blockIdx.y;
  const int bb = bh / HN, h = bh % HN;
  const size_t base = (size_t)bh * SN * DKN;
  const float scale = 0.125f;  // 64^-0.5

  // Q fragments: [query tile][dk chunk], resident for the whole loop.
  Frag aq[2][2];
#pragma unroll
  for (int qt = 0; qt < 2; ++qt)
#pragma unroll
    for (int c = 0; c < 2; ++c)
      aq[qt][c] = loadA(q + base + (size_t)(s0 + qt * 16 + r) * DKN + c * 32, hi);

  float m_run[2][8], l_run[2][8];
  v8f acco[2][4];
#pragma unroll
  for (int qt = 0; qt < 2; ++qt) {
#pragma unroll
    for (int i = 0; i < 8; ++i) { m_run[qt][i] = -1e30f; l_run[qt][i] = 0.0f; }
#pragma unroll
    for (int c = 0; c < 4; ++c) acco[qt][c] = v8f{};
  }

  for (int j0 = 0; j0 < SN; j0 += 32) {
    // prefetch next key block (global_prefetch_b8 path)
    __builtin_prefetch(k + base + (size_t)(j0 + 32 + r) * DKN, 0, 3);
    __builtin_prefetch(vT + base + (size_t)(2 * r) * SN + j0 + 32, 0, 3);

    // ---- shared K fragments: [key sub-tile][dk chunk] ----
    Frag bk[2][2];
#pragma unroll
    for (int sub = 0; sub < 2; ++sub)
#pragma unroll
      for (int c = 0; c < 2; ++c)
        bk[sub][c] =
            loadB(k + base + (size_t)(j0 + sub * 16 + r) * DKN + c * 32, hi);

#pragma unroll
    for (int qt = 0; qt < 2; ++qt) {
      // ---- S tile: 16 queries x 32 keys ----
      float sv[2][8];
#pragma unroll
      for (int sub = 0; sub < 2; ++sub) {
        v8f accs = {};
        accs = wmma_bf16(aq[qt][0], bk[sub][0], accs);
        accs = wmma_bf16(aq[qt][1], bk[sub][1], accs);
#pragma unroll
        for (int i = 0; i < 8; ++i) sv[sub][i] = accs[i] * scale;
      }
      // ---- online softmax (reductions stay inside 16-lane row groups) ----
      float mnew[8], rs[8];
#pragma unroll
      for (int i = 0; i < 8; ++i) {
        float t = fmaxf(sv[0][i], sv[1][i]);
        for (int off = 1; off < 16; off <<= 1)
          t = fmaxf(t, __shfl_xor(t, off, 32));
        mnew[i] = fmaxf(m_run[qt][i], t);
      }
#pragma unroll
      for (int i = 0; i < 8; ++i) {
        const float p0 = __expf(sv[0][i] - mnew[i]);
        const float p1 = __expf(sv[1][i] - mnew[i]);
        sv[0][i] = p0; sv[1][i] = p1;
        float t = p0 + p1;
        for (int off = 1; off < 16; off <<= 1) t += __shfl_xor(t, off, 32);
        rs[i] = t;
      }
#pragma unroll
      for (int i = 0; i < 8; ++i) {
        const float alpha = __expf(m_run[qt][i] - mnew[i]);
        l_run[qt][i] = l_run[qt][i] * alpha + rs[i];
        m_run[qt][i] = mnew[i];
#pragma unroll
        for (int c = 0; c < 4; ++c) acco[qt][c][i] *= alpha;
      }
      // ---- restage P (C-layout) into A-fragment layout via LDS ----
#pragma unroll
      for (int sub = 0; sub < 2; ++sub)
#pragma unroll
        for (int i = 0; i < 8; ++i)
          p_lds[qt][(i + hi * 8) * 32 + sub * 16 + r] = f2bf(sv[sub][i]);
    }

    // ---- shared V fragments (vT is [d][s]: K-contiguous) ----
    Frag bv[4];
#pragma unroll
    for (int c = 0; c < 4; ++c)
      bv[c] = loadB(vT + base + (size_t)(c * 16 + r) * SN + j0, hi);

#pragma unroll
    for (int qt = 0; qt < 2; ++qt) {
      Frag ap = loadA(&p_lds[qt][r * 32], hi);  // two ds_load_b128
#pragma unroll
      for (int c = 0; c < 4; ++c) acco[qt][c] = wmma_bf16(ap, bv[c], acco[qt][c]);
    }
  }

  // ---- normalize and store res[b][s][h*64 + d] (bf16) ----
#pragma unroll
  for (int qt = 0; qt < 2; ++qt)
#pragma unroll
    for (int c = 0; c < 4; ++c)
#pragma unroll
      for (int i = 0; i < 8; ++i) {
        const int s = s0 + qt * 16 + i + hi * 8;
        res[((size_t)bb * SN + s) * HDN + h * DKN + c * 16 + r] =
            f2bf(acco[qt][c][i] / l_run[qt][i]);
      }
}

// ---------------------------------------------------------------------------
// Kernel 3: out[b,c,s] = sum_d res[b,s,d]*w_out[c,d] + b_out[c] + x[b,c,s]
// 16(s) x 32(c) per wave.
// ---------------------------------------------------------------------------
__global__ __launch_bounds__(32) void outproj_kernel(
    const unsigned short* __restrict__ res, const unsigned short* __restrict__ wob,
    const float* __restrict__ b_out, const float* __restrict__ x,
    float* __restrict__ out) {
  const int lane = threadIdx.x, hi = lane >> 4, r = lane & 15;
  const int s0 = blockIdx.x * 16, c0 = blockIdx.y * 32, bb = blockIdx.z;

  const unsigned short* arow = res + ((size_t)bb * SN + s0 + r) * HDN;
  const unsigned short* brow0 = wob + (size_t)(c0 + r) * HDN;
  const unsigned short* brow1 = wob + (size_t)(c0 + 16 + r) * HDN;

  v8f acc0 = {}, acc1 = {};
  for (int kk = 0; kk < HDN; kk += 32) {
    Frag a  = loadA(arow + kk, hi);
    Frag b0 = loadB(brow0 + kk, hi);
    Frag b1 = loadB(brow1 + kk, hi);
    acc0 = wmma_bf16(a, b0, acc0);
    acc1 = wmma_bf16(a, b1, acc1);
  }
#pragma unroll
  for (int ct = 0; ct < 2; ++ct) {
    const v8f acc = ct ? acc1 : acc0;
    const int c = c0 + ct * 16 + r;
#pragma unroll
    for (int i = 0; i < 8; ++i) {
      const int s = s0 + i + hi * 8;
      const size_t idx = ((size_t)bb * CN + c) * SN + s;
      out[idx] = acc[i] + b_out[c] + x[idx];
    }
  }
}

// ---------------------------------------------------------------------------
extern "C" void kernel_launch(void* const* d_in, const int* in_sizes, int n_in,
                              void* d_out, int out_size, void* d_ws, size_t ws_size,
                              hipStream_t stream) {
  (void)in_sizes; (void)n_in; (void)out_size; (void)ws_size;
  const float* x      = (const float*)d_in[0];
  const float* w_proj = (const float*)d_in[1];
  const float* b_proj = (const float*)d_in[2];
  const float* w_out  = (const float*)d_in[3];
  const float* b_out  = (const float*)d_in[4];
  float* out = (float*)d_out;

  // bf16 workspace layout (ushort elems); every region 16B-aligned.
  unsigned short* p = (unsigned short*)d_ws;
  const size_t n_x   = (size_t)BN * SN * CN;        // xbf  [b][s][c]
  const size_t n_wp  = (size_t)ON * CN;             // wpb
  const size_t n_wo  = (size_t)CN * HDN;            // wob
  const size_t n_qkv = (size_t)BN * HN * SN * DKN;  // q / k / vT
  unsigned short* xbf  = p;             p += n_x;
  unsigned short* wpb  = p;             p += n_wp;
  unsigned short* wob  = p;             p += n_wo;
  unsigned short* qw   = p;             p += n_qkv;
  unsigned short* kw   = p;             p += n_qkv;
  unsigned short* vtw  = p;             p += n_qkv;
  unsigned short* resw = p;             /* B*S*HD */

  cvt_kernel<<<dim3((unsigned)((n_wp + 255) / 256)), 256, 0, stream>>>(
      w_proj, wpb, (int)n_wp);
  cvt_kernel<<<dim3((unsigned)((n_wo + 255) / 256)), 256, 0, stream>>>(
      w_out, wob, (int)n_wo);
  xt_kernel<<<dim3(SN / 32, CN / 32, BN), dim3(32, 8), 0, stream>>>(x, xbf);

  qkv_kernel<<<dim3(SN / 16, ON / 32, BN), 32, 0, stream>>>(xbf, wpb, b_proj,
                                                            qw, kw, vtw);
  attn_kernel<<<dim3(SN / 32, BN * HN), 32, 0, stream>>>(qw, kw, vtw, resw);
  outproj_kernel<<<dim3(SN / 16, CN / 32, BN), 32, 0, stream>>>(resw, wob,
                                                                b_out, x, out);
}
